// VectorQuantizer_7739531067664
// MI455X (gfx1250) — compile-verified, matrix operand order checked by reading, not run on hardware
//
#include <hip/hip_runtime.h>
#include <cstdint>
#include <cmath>

typedef __attribute__((ext_vector_type(16))) __bf16 v16bf;
typedef __attribute__((ext_vector_type(8)))  float  v8f;

#define NCODE   4096
#define EDIM    256
#define NROWS   32768        // 32 * 32 * 32  (b * h * w)
#define ZB      262144       // per-batch stride in floats (256*32*32)

__device__ __forceinline__ __bf16 f2bf(float f) { return (__bf16)f; }

// ---------------------------------------------------------------------------
// Kernel 1: codebook norms ||c_j||^2 ; also zero histogram
// grid 512 x 256 (8 waves/block, one codebook row per wave)
// ---------------------------------------------------------------------------
__global__ void vq_cnorm(const float* __restrict__ cbk,
                         float* __restrict__ cnorm,
                         unsigned* __restrict__ hist) {
    int lane = threadIdx.x & 31;
    int wave = threadIdx.x >> 5;
    int j = blockIdx.x * 8 + wave;
    const float* row = cbk + (size_t)j * EDIM;
    float s = 0.f;
#pragma unroll
    for (int q = 0; q < EDIM / 32; ++q) {
        float v = row[lane + q * 32];
        s += v * v;
    }
#pragma unroll
    for (int off = 16; off >= 1; off >>= 1) s += __shfl_xor(s, off, 32);
    if (lane == 0) cnorm[j] = s;
    int g = blockIdx.x * 256 + threadIdx.x;
    if (g < NCODE) hist[g] = 0u;
}

// ---------------------------------------------------------------------------
// Kernel 2: WMMA distance GEMM + argmin.
// grid 256 blocks x 256 threads. Each block: 8 waves x 16 rows = 128 rows.
// A (z rows) kept in registers as bf16 hi/lo fragments; codebook chunks of 64
// codes staged through LDS as pre-split bf16 hi/lo, shared by all waves.
// ---------------------------------------------------------------------------
__global__ void __launch_bounds__(256, 1)
vq_argmin(const float* __restrict__ z,
          const float* __restrict__ cbk,
          const float* __restrict__ cnorm,
          int* __restrict__ idx_out,
          float* __restrict__ idx_f_out,
          unsigned* __restrict__ hist) {
    __shared__ __bf16 sBh[64][EDIM];   // 32 KB
    __shared__ __bf16 sBl[64][EDIM];   // 32 KB

    const int lane = threadIdx.x & 31;
    const int wave = threadIdx.x >> 5;
    const int m    = lane & 15;
    const int half = lane >> 4;

    // ---- load A stripe: rows r0..r0+15, all K=256, split hi/lo bf16 ----
    const int r = blockIdx.x * 128 + wave * 16 + m;   // global row (b*1024+hw)
    const float* zbase = z + (size_t)(r >> 10) * ZB + (r & 1023);

    v16bf a_hi[8], a_lo[8];
#pragma unroll
    for (int f = 0; f < 8; ++f) {
#pragma unroll
        for (int t = 0; t < 16; ++t) {
            int k = f * 32 + half * 8 + (t & 7) + ((t >> 3) << 4);
            float v = zbase[(size_t)k * 1024];
            __bf16 h = f2bf(v);
            a_hi[f][t] = h;
            a_lo[f][t] = f2bf(v - (float)h);
        }
    }

    float minv[8];
    int   mini[8];
#pragma unroll
    for (int v = 0; v < 8; ++v) { minv[v] = 3.4e38f; mini[v] = 0; }

    for (int chunk = 0; chunk < NCODE / 64; ++chunk) {
        const int jbase = chunk * 64;
        __syncthreads();
        // cooperative fill: 64 codes x 256 elems = 16384, 64 per thread
#pragma unroll 4
        for (int q = 0; q < 64; ++q) {
            int e  = threadIdx.x + q * 256;
            int jj = e >> 8;
            int kk = e & 255;
            float v = cbk[(size_t)(jbase + jj) * EDIM + kk];
            __bf16 h = f2bf(v);
            sBh[jj][kk] = h;
            sBl[jj][kk] = f2bf(v - (float)h);
        }
        __syncthreads();

#pragma unroll
        for (int ct = 0; ct < 4; ++ct) {           // four 16-col tiles
            const int n = m;                       // B lane -> column
            const int j = jbase + ct * 16 + n;
            v8f acc = {};
#pragma unroll
            for (int f = 0; f < 8; ++f) {
                const __bf16* ph = &sBh[ct * 16 + n][f * 32 + half * 8];
                const __bf16* pl = &sBl[ct * 16 + n][f * 32 + half * 8];
                v16bf bh, bl;
#pragma unroll
                for (int t = 0; t < 8; ++t) { bh[t] = ph[t];      bl[t] = pl[t]; }
#pragma unroll
                for (int t = 0; t < 8; ++t) { bh[8 + t] = ph[16 + t]; bl[8 + t] = pl[16 + t]; }
                acc = __builtin_amdgcn_wmma_f32_16x16x32_bf16(false, a_hi[f], false, bh,
                                                              (short)0, acc, false, false);
                acc = __builtin_amdgcn_wmma_f32_16x16x32_bf16(false, a_hi[f], false, bl,
                                                              (short)0, acc, false, false);
                acc = __builtin_amdgcn_wmma_f32_16x16x32_bf16(false, a_lo[f], false, bh,
                                                              (short)0, acc, false, false);
            }
            const float cn = cnorm[j];
#pragma unroll
            for (int v = 0; v < 8; ++v) {
                float d = cn - 2.0f * acc[v];      // + ||z||^2 omitted (row-const)
                if (d < minv[v]) { minv[v] = d; mini[v] = j; }
            }
        }
    }

    // ---- per-row argmin across the 16 lanes of each half ----
#pragma unroll
    for (int v = 0; v < 8; ++v) {
        float mv = minv[v];
        int   mi = mini[v];
#pragma unroll
        for (int off = 8; off >= 1; off >>= 1) {
            float ov = __shfl_xor(mv, off, 16);
            int   oi = __shfl_xor(mi, off, 16);
            if (ov < mv || (ov == mv && oi < mi)) { mv = ov; mi = oi; }
        }
        if (m == 0) {
            int row = blockIdx.x * 128 + wave * 16 + v + 8 * half;
            idx_out[row]   = mi;
            idx_f_out[row] = (float)mi;
            atomicAdd(&hist[mi], 1u);
        }
    }
}

// ---------------------------------------------------------------------------
// Kernel 3: z_q gather -> NCHW output + deterministic loss partials
// grid 32768 x 256 ; block = fixed channel e, 256 consecutive rows i
// ---------------------------------------------------------------------------
__global__ void vq_gather(const float* __restrict__ z,
                          const float* __restrict__ cbk,
                          const int* __restrict__ idx,
                          float* __restrict__ zq_out,
                          float* __restrict__ partials) {
    int e = blockIdx.x >> 7;                              // 0..255
    int i = ((blockIdx.x & 127) << 8) + threadIdx.x;      // 0..32767
    int b = i >> 10, hw = i & 1023;
    size_t off = (size_t)b * ZB + (size_t)e * 1024 + hw;
    float zv = z[off];
    float q  = cbk[(size_t)idx[i] * EDIM + e];
    zq_out[off] = q;                                      // z_q_st == z_q numerically
    float d = q - zv;
    __shared__ float red[256];
    red[threadIdx.x] = d * d;
    __syncthreads();
#pragma unroll
    for (int s = 128; s > 0; s >>= 1) {
        if (threadIdx.x < s) red[threadIdx.x] += red[threadIdx.x + s];
        __syncthreads();
    }
    if (threadIdx.x == 0) partials[blockIdx.x] = red[0];
}

// ---------------------------------------------------------------------------
// Kernel 4: fused zero-fill + scatter of one-hot encodings (float2 stores;
// d_out one-hot section is only 8-byte aligned). grid 262144 x 256.
// ---------------------------------------------------------------------------
__global__ void vq_onehot(const int* __restrict__ idx, float2* __restrict__ out) {
    size_t p = (size_t)blockIdx.x * 256 + threadIdx.x;    // float2 index
    int i    = (int)(p >> 11);                            // row (2048 float2/row)
    int col2 = (int)(p & 2047);
    int id   = idx[i];
    float2 v = make_float2(0.f, 0.f);
    if ((id >> 1) == col2) {
        if (id & 1) v.y = 1.f; else v.x = 1.f;
    }
    out[p] = v;
}

// ---------------------------------------------------------------------------
// Kernel 5: finalize loss + perplexity (single block, deterministic)
// ---------------------------------------------------------------------------
__global__ void vq_finalize(const float* __restrict__ partials,
                            const unsigned* __restrict__ hist,
                            float* __restrict__ loss_out,
                            float* __restrict__ perp_out) {
    __shared__ float red[256];
    float s = 0.f;
    for (int q = threadIdx.x; q < NROWS; q += 256) s += partials[q];
    red[threadIdx.x] = s;
    __syncthreads();
#pragma unroll
    for (int t = 128; t > 0; t >>= 1) {
        if (threadIdx.x < t) red[threadIdx.x] += red[threadIdx.x + t];
        __syncthreads();
    }
    if (threadIdx.x == 0)
        loss_out[0] = 1.25f * red[0] / 8388608.0f;        // (1 + BETA) * MSE
    __syncthreads();
    float h = 0.f;
    for (int q = threadIdx.x; q < NCODE; q += 256) {
        float em = (float)hist[q] * (1.0f / 32768.0f);
        h += em * logf(em + 1e-10f);
    }
    red[threadIdx.x] = h;
    __syncthreads();
#pragma unroll
    for (int t = 128; t > 0; t >>= 1) {
        if (threadIdx.x < t) red[threadIdx.x] += red[threadIdx.x + t];
        __syncthreads();
    }
    if (threadIdx.x == 0) perp_out[0] = expf(-red[0]);
}

// ---------------------------------------------------------------------------
extern "C" void kernel_launch(void* const* d_in, const int* in_sizes, int n_in,
                              void* d_out, int out_size, void* d_ws, size_t ws_size,
                              hipStream_t stream) {
    const float* z   = (const float*)d_in[0];   // (32,256,32,32)
    const float* cbk = (const float*)d_in[1];   // (4096,256)
    float* out = (float*)d_out;

    // output layout (floats): loss | z_q_out | perplexity | one_hot | indices
    const size_t OFF_LOSS = 0;
    const size_t OFF_ZQ   = 1;
    const size_t OFF_PERP = OFF_ZQ + (size_t)32 * 256 * 32 * 32;     // 8388609
    const size_t OFF_OH   = OFF_PERP + 1;                            // 8388610
    const size_t OFF_IDX  = OFF_OH + (size_t)NROWS * NCODE;          // 142606338

    // workspace layout (bytes)
    char* ws = (char*)d_ws;
    int*      idx_i    = (int*)(ws + 0);            // 32768 * 4
    float*    cnorm    = (float*)(ws + 131072);     // 4096 * 4
    unsigned* hist     = (unsigned*)(ws + 147456);  // 4096 * 4
    float*    partials = (float*)(ws + 163840);     // 32768 * 4

    vq_cnorm<<<512, 256, 0, stream>>>(cbk, cnorm, hist);
    vq_argmin<<<256, 256, 0, stream>>>(z, cbk, cnorm, idx_i, out + OFF_IDX, hist);
    vq_gather<<<32768, 256, 0, stream>>>(z, cbk, idx_i, out + OFF_ZQ, partials);
    vq_onehot<<<262144, 256, 0, stream>>>(idx_i, (float2*)(out + OFF_OH));
    vq_finalize<<<1, 256, 0, stream>>>(partials, hist, out + OFF_LOSS, out + OFF_PERP);
}
